// SimpleNet_37555194037048
// MI455X (gfx1250) — matrix operand order repaired and checked by reading, not compiled, not run on hardware
//
#include <hip/hip_runtime.h>
#include <math.h>

typedef __attribute__((ext_vector_type(16))) __bf16 v16bf;
typedef __attribute__((ext_vector_type(8)))  float  v8f;

#define NNODES 50000
#define HDIM 128

static __device__ __forceinline__ unsigned short f2bf(float f) {
    union { float f; unsigned int u; } x; x.f = f;
    unsigned int u = x.u;
    unsigned int r = u + 0x7FFFu + ((u >> 16) & 1u);   // round-to-nearest-even
    return (unsigned short)(r >> 16);
}

static __device__ __forceinline__ void atomAddF(float* p, float v) {
    __hip_atomic_fetch_add(p, v, __ATOMIC_RELAXED, __HIP_MEMORY_SCOPE_AGENT);
}

// ---------------------------------------------------------------------------
// Pack a f32 [K x 128] weight into wave32 WMMA bf16 B-fragment order:
// P[((ntile*ksteps + ks)*32 + lane)*16 + j] = W[k][n]
//   n = ntile*16 + lane%16 ; k = ks*32 + (lane<16 ? 0 : 8) + (j<8 ? j : j+8)  (0 if k>=K)
// ---------------------------------------------------------------------------
__global__ void pack_weight_bf16(const float* __restrict__ W, unsigned short* __restrict__ P,
                                 int K, int ksteps, int total) {
    int p = blockIdx.x * blockDim.x + threadIdx.x;
    if (p >= total) return;
    int j     = p & 15;
    int lane  = (p >> 4) & 31;
    int ks    = (p >> 9) % ksteps;
    int ntile = (p >> 9) / ksteps;
    int n = ntile * 16 + (lane & 15);
    int k = ks * 32 + ((lane < 16) ? 0 : 8) + ((j < 8) ? j : j + 8);
    float v = (k < K) ? W[(size_t)k * HDIM + n] : 0.0f;
    P[p] = f2bf(v);
}

// Build one half-row A fragment chunk (8 f32 -> 8 bf16), zero if !ok
static __device__ __forceinline__ void loadA8(const float* __restrict__ p, bool ok,
                                              unsigned short* __restrict__ dst) {
    if (ok) {
        float4 p0 = *(const float4*)(p);
        float4 p1 = *(const float4*)(p + 4);
        dst[0] = f2bf(p0.x); dst[1] = f2bf(p0.y); dst[2] = f2bf(p0.z); dst[3] = f2bf(p0.w);
        dst[4] = f2bf(p1.x); dst[5] = f2bf(p1.y); dst[6] = f2bf(p1.z); dst[7] = f2bf(p1.w);
    } else {
#pragma unroll
        for (int j = 0; j < 8; ++j) dst[j] = 0;
    }
}

// ---------------------------------------------------------------------------
// WMMA GEMM: C[M x 128] = act( A[M x K] * B + bias ), optional row gather on A.
// One wave computes a 16x64 tile: one A fragment feeds 4 WMMAs per K-step.
// ---------------------------------------------------------------------------
template <int KSTEPS>
__global__ void __launch_bounds__(128) gemm_bf16_wmma(
    const float* __restrict__ A, int lda,
    const unsigned short* __restrict__ Bp,
    const float* __restrict__ bias,
    float* __restrict__ C, int ldc,
    int M, int K,
    const int* __restrict__ gather, int relu)
{
    const int wave = threadIdx.x >> 5;
    const int lane = threadIdx.x & 31;
    const int mtiles = (M + 15) >> 4;
    const int wgs = mtiles * 2;               // 2 column-quadrants (64 cols) per m-tile
    int wg = blockIdx.x * 4 + wave;
    if (wg >= wgs) return;                    // wave-uniform exit (EXEC all-1 for WMMA)
    const int mtile = wg >> 1;
    const int nq    = wg & 1;                 // columns nq*64 .. nq*64+63
    const bool fullTile = ((mtile + 1) * 16 <= M);

    const int rlocal = lane & 15;
    const int m = mtile * 16 + rlocal;        // A row this lane supplies
    const bool rowOK = fullTile || (m < M);
    long arow = 0;
    if (rowOK) arow = gather ? (long)gather[m] : (long)m;
    const float* Arow = A + (size_t)arow * lda;

    const int n0 = nq * 64 + rlocal;          // output column (first of 4 tiles)
    v8f c0, c1, c2, c3;
    {
        float b0 = bias[n0], b1 = bias[n0 + 16], b2 = bias[n0 + 32], b3 = bias[n0 + 48];
#pragma unroll
        for (int i = 0; i < 8; ++i) { c0[i] = b0; c1[i] = b1; c2[i] = b2; c3[i] = b3; }
    }

    // per-lane B base: 4 consecutive ntiles starting at nq*4
    const unsigned short* Bq = Bp + (size_t)(nq * 4) * KSTEPS * 512 + (size_t)lane * 16;

    const int khalf = (lane >= 16) ? 8 : 0;
#pragma unroll
    for (int ks = 0; ks < KSTEPS; ++ks) {
        union { v16bf v; unsigned short u[16]; } a;
        const int k0 = ks * 32 + khalf;       // elements 0..7  <- K = k0..k0+7
        const int k1 = k0 + 16;               // elements 8..15 <- K = k0+16..k0+23
        loadA8(Arow + k0, rowOK && (k0 + 8) <= K, a.u);
        loadA8(Arow + k1, rowOK && (k1 + 8) <= K, a.u + 8);
        v16bf b0 = *reinterpret_cast<const v16bf*>(Bq + (size_t)(0 * KSTEPS + ks) * 512);
        v16bf b1 = *reinterpret_cast<const v16bf*>(Bq + (size_t)(1 * KSTEPS + ks) * 512);
        v16bf b2 = *reinterpret_cast<const v16bf*>(Bq + (size_t)(2 * KSTEPS + ks) * 512);
        v16bf b3 = *reinterpret_cast<const v16bf*>(Bq + (size_t)(3 * KSTEPS + ks) * 512);
        c0 = __builtin_amdgcn_wmma_f32_16x16x32_bf16(false, a.v, false, b0, (short)0, c0, false, false);
        c1 = __builtin_amdgcn_wmma_f32_16x16x32_bf16(false, a.v, false, b1, (short)0, c1, false, false);
        c2 = __builtin_amdgcn_wmma_f32_16x16x32_bf16(false, a.v, false, b2, (short)0, c2, false, false);
        c3 = __builtin_amdgcn_wmma_f32_16x16x32_bf16(false, a.v, false, b3, (short)0, c3, false, false);
    }

    const int mrow0 = mtile * 16 + ((lane >= 16) ? 8 : 0);
    if (fullTile) {
#pragma unroll
        for (int v = 0; v < 8; ++v) {
            float* crow = C + (size_t)(mrow0 + v) * ldc + n0;
            float r0 = c0[v], r1 = c1[v], r2 = c2[v], r3 = c3[v];
            if (relu) { r0 = fmaxf(r0, 0.f); r1 = fmaxf(r1, 0.f); r2 = fmaxf(r2, 0.f); r3 = fmaxf(r3, 0.f); }
            crow[0] = r0; crow[16] = r1; crow[32] = r2; crow[48] = r3;
        }
    } else {
#pragma unroll
        for (int v = 0; v < 8; ++v) {
            int mm = mrow0 + v;
            if (mm < M) {
                float* crow = C + (size_t)mm * ldc + n0;
                float r0 = c0[v], r1 = c1[v], r2 = c2[v], r3 = c3[v];
                if (relu) { r0 = fmaxf(r0, 0.f); r1 = fmaxf(r1, 0.f); r2 = fmaxf(r2, 0.f); r3 = fmaxf(r3, 0.f); }
                crow[0] = r0; crow[16] = r1; crow[32] = r2; crow[48] = r3;
            }
        }
    }
}

// ---------------------------------------------------------------------------
// Graph aggregation: one wave per edge; msg = relu(h[src]); agg[tgt] += msg
// ---------------------------------------------------------------------------
__global__ void __launch_bounds__(256) edge_scatter(
    const float* __restrict__ h, int lda,
    const int* __restrict__ src, const int* __restrict__ tgt,
    float* __restrict__ agg, float* __restrict__ cnt, int E)
{
    int e = blockIdx.x * 8 + (threadIdx.x >> 5);
    if (e >= E) return;
    int lane = threadIdx.x & 31;
    int s = src[e];
    int t = tgt[e];
    float4 v = *(const float4*)(h + (size_t)s * lda + lane * 4);
    v.x = fmaxf(v.x, 0.f); v.y = fmaxf(v.y, 0.f);
    v.z = fmaxf(v.z, 0.f); v.w = fmaxf(v.w, 0.f);
    float* ap = agg + (size_t)t * HDIM + lane * 4;
    atomAddF(ap + 0, v.x); atomAddF(ap + 1, v.y);
    atomAddF(ap + 2, v.z); atomAddF(ap + 3, v.w);
    if (lane == 0) atomAddF(cnt + t, 1.0f);
}

// agg = (1+eps)*h + agg / max(cnt,1)   (in place on agg)
__global__ void combine_eps(const float* __restrict__ h, int lda,
                            const float* __restrict__ epsp, const float* __restrict__ cnt,
                            float* __restrict__ agg, int total4)
{
    int idx = blockIdx.x * blockDim.x + threadIdx.x;
    if (idx >= total4) return;
    int i  = idx >> 5;
    int c4 = (idx & 31) * 4;
    float inv   = 1.0f / fmaxf(cnt[i], 1.0f);
    float onepe = 1.0f + epsp[0];
    float4 hv = *(const float4*)(h + (size_t)i * lda + c4);
    float4 av = *(float4*)(agg + (size_t)i * HDIM + c4);
    av.x = onepe * hv.x + av.x * inv;
    av.y = onepe * hv.y + av.y * inv;
    av.z = onepe * hv.z + av.z * inv;
    av.w = onepe * hv.w + av.w * inv;
    *(float4*)(agg + (size_t)i * HDIM + c4) = av;
}

// Column-wise sum / sumsq for batchnorm (biased stats), 128 threads = 128 cols
__global__ void bn_stats(const float* __restrict__ x, int lda,
                         float* __restrict__ stats, int Nrows)
{
    int c = threadIdx.x;
    float s = 0.f, s2 = 0.f;
    for (int i = blockIdx.x; i < Nrows; i += gridDim.x) {
        float v = x[(size_t)i * lda + c];
        s += v; s2 += v * v;
    }
    atomAddF(&stats[c], s);
    atomAddF(&stats[HDIM + c], s2);
}

__global__ void bn_apply(float* __restrict__ x, int lda, const float* __restrict__ stats,
                         const float* __restrict__ gamma, const float* __restrict__ beta,
                         int Nrows)
{
    int idx = blockIdx.x * blockDim.x + threadIdx.x;
    if (idx >= Nrows * HDIM) return;
    int i = idx >> 7;
    int c = idx & 127;
    float invN = 1.0f / (float)Nrows;
    float m = stats[c] * invN;
    float v = stats[HDIM + c] * invN - m * m;
    float inv = rsqrtf(v + 1e-5f);
    size_t o = (size_t)i * lda + c;
    x[o] = (x[o] - m) * inv * gamma[c] + beta[c];
}

__global__ void fillf(float* __restrict__ p, float v, int n) {
    int i = blockIdx.x * blockDim.x + threadIdx.x;
    if (i < n) p[i] = v;
}

// logits = X @ w4[128x2] + b4 ; out = log_softmax(logits)
__global__ void final_head(const float* __restrict__ X, const float* __restrict__ w4,
                           const float* __restrict__ b4, float* __restrict__ out, int M)
{
    int r = blockIdx.x * blockDim.x + threadIdx.x;
    if (r >= M) return;
    const float* x = X + (size_t)r * HDIM;
    float z0 = b4[0], z1 = b4[1];
#pragma unroll 4
    for (int k = 0; k < HDIM; ++k) {
        float v = x[k];
        z0 += v * w4[2 * k];
        z1 += v * w4[2 * k + 1];
    }
    float mx = fmaxf(z0, z1);
    float l = logf(expf(z0 - mx) + expf(z1 - mx));
    out[2 * r]     = z0 - mx - l;
    out[2 * r + 1] = z1 - mx - l;
}

// ---------------------------------------------------------------------------
extern "C" void kernel_launch(void* const* d_in, const int* in_sizes, int n_in,
                              void* d_out, int out_size, void* d_ws, size_t ws_size,
                              hipStream_t stream)
{
    const int N = NNODES;
    const int E = in_sizes[1] / 2;
    const int M = in_sizes[3];

    const float* X    = (const float*)d_in[0];
    const int*   ei1  = (const int*)d_in[1];
    const int*   ei2  = (const int*)d_in[2];
    const int*   idxs = (const int*)d_in[3];

    // params: enc{w1,b1,w2,b2} convs[8]{w1,b1,w2,b2,gamma,beta,eps} joint{w1,b1,w2,b2} lin{w1..b4}
    const float* enc_w1 = (const float*)d_in[4];
    const float* enc_b1 = (const float*)d_in[5];
    const float* enc_w2 = (const float*)d_in[6];
    const float* enc_b2 = (const float*)d_in[7];
    auto cvp = [&](int ci, int f) { return (const float*)d_in[8 + ci * 7 + f]; };
    const int JB = 8 + 8 * 7;   // 64
    const float* j_w1 = (const float*)d_in[JB + 0];
    const float* j_b1 = (const float*)d_in[JB + 1];
    const float* j_w2 = (const float*)d_in[JB + 2];
    const float* j_b2 = (const float*)d_in[JB + 3];
    const float* l_w1 = (const float*)d_in[JB + 4];
    const float* l_b1 = (const float*)d_in[JB + 5];
    const float* l_w2 = (const float*)d_in[JB + 6];
    const float* l_b2 = (const float*)d_in[JB + 7];
    const float* l_w3 = (const float*)d_in[JB + 8];
    const float* l_b3 = (const float*)d_in[JB + 9];
    const float* l_w4 = (const float*)d_in[JB + 10];
    const float* l_b4 = (const float*)d_in[JB + 11];

    // workspace layout (float offsets)
    float* ws = (float*)d_ws;
    const size_t OFF_FEATS = 0;                                  // N x 640
    const size_t OFF_XCAT  = OFF_FEATS + (size_t)N * 640;        // N x 256
    const size_t OFF_TMP1  = OFF_XCAT  + (size_t)N * 256;        // N x 128
    const size_t OFF_AGG   = OFF_TMP1  + (size_t)N * 128;        // N x 128
    const size_t OFF_CNT   = OFF_AGG   + (size_t)N * 128;        // N
    const size_t OFF_STATS = OFF_CNT   + (size_t)N;              // 256
    const size_t OFF_WB    = ((OFF_STATS + 256 + 63) / 64) * 64; // bf16 packed weights
    float* FEATS = ws + OFF_FEATS;
    float* XCAT  = ws + OFF_XCAT;
    float* TMP1  = ws + OFF_TMP1;
    float* AGG   = ws + OFF_AGG;
    float* CNT   = ws + OFF_CNT;
    float* STATS = ws + OFF_STATS;
    unsigned short* WB = (unsigned short*)(ws + OFF_WB);

    // packed-weight element offsets (512 * ksteps * 8 per matrix)
    const size_t P_ENC1 = 0;            // K=8   -> 4096
    const size_t P_ENC2 = 4096;         // K=128 -> 16384
    const size_t P_CONV = 20480;        // 16 x 16384
    const size_t P_J1   = 282624;       // K=256 -> 32768
    const size_t P_J2   = 315392;       // 16384
    const size_t P_L1   = 331776;       // K=640 -> 81920
    const size_t P_L2   = 413696;       // 16384
    const size_t P_L3   = 430080;       // 16384

    auto packW = [&](const float* W, unsigned short* P, int K) {
        int ksteps = (K + 31) / 32;
        int total = 8 * ksteps * 512;
        pack_weight_bf16<<<(total + 255) / 256, 256, 0, stream>>>(W, P, K, ksteps, total);
    };
    packW(enc_w1, WB + P_ENC1, 8);
    packW(enc_w2, WB + P_ENC2, 128);
    for (int ci = 0; ci < 8; ++ci) {
        packW(cvp(ci, 0), WB + P_CONV + (size_t)ci * 32768, 128);
        packW(cvp(ci, 2), WB + P_CONV + (size_t)ci * 32768 + 16384, 128);
    }
    packW(j_w1, WB + P_J1, 256);
    packW(j_w2, WB + P_J2, 128);
    packW(l_w1, WB + P_L1, 640);
    packW(l_w2, WB + P_L2, 128);
    packW(l_w3, WB + P_L3, 128);

    auto gemm = [&](const float* A, int lda, const unsigned short* Bp, const float* bias,
                    float* C, int ldc, int Mr, int K, const int* gather, int relu) {
        int mtiles = (Mr + 15) / 16;
        int blocks = (mtiles * 2 + 3) / 4;
        switch ((K + 31) / 32) {
        case 1:  gemm_bf16_wmma<1><<<blocks, 128, 0, stream>>>(A, lda, Bp, bias, C, ldc, Mr, K, gather, relu); break;
        case 4:  gemm_bf16_wmma<4><<<blocks, 128, 0, stream>>>(A, lda, Bp, bias, C, ldc, Mr, K, gather, relu); break;
        case 8:  gemm_bf16_wmma<8><<<blocks, 128, 0, stream>>>(A, lda, Bp, bias, C, ldc, Mr, K, gather, relu); break;
        case 20: gemm_bf16_wmma<20><<<blocks, 128, 0, stream>>>(A, lda, Bp, bias, C, ldc, Mr, K, gather, relu); break;
        default: break;
        }
    };

    // encoder: h0 = relu(X @ w1 + b1) @ w2 + b2   -> FEATS block 0
    gemm(X, 8, WB + P_ENC1, enc_b1, TMP1, 128, N, 8, nullptr, 1);
    gemm(TMP1, 128, WB + P_ENC2, enc_b2, FEATS, 640, N, 128, nullptr, 0);

    for (int st = 0; st < 4; ++st) {
        const float* h = FEATS + st * 128;     // lda 640
        for (int g = 0; g < 2; ++g) {
            int ci = st * 2 + g;
            const int* ei = (g == 0) ? ei1 : ei2;
            int nz = N * 128 + N;
            fillf<<<(nz + 255) / 256, 256, 0, stream>>>(AGG, 0.0f, nz);
            edge_scatter<<<(E + 7) / 8, 256, 0, stream>>>(h, 640, ei, ei + E, AGG, CNT, E);
            combine_eps<<<(N * 32 + 255) / 256, 256, 0, stream>>>(h, 640, cvp(ci, 6), CNT, AGG, N * 32);
            gemm(AGG, 128, WB + P_CONV + (size_t)ci * 32768, cvp(ci, 1), TMP1, 128, N, 128, nullptr, 1);
            float* xg = XCAT + g * 128;        // ldc 256 -> [x1 | x2] contiguous
            gemm(TMP1, 128, WB + P_CONV + (size_t)ci * 32768 + 16384, cvp(ci, 3), xg, 256, N, 128, nullptr, 1);
            fillf<<<1, 256, 0, stream>>>(STATS, 0.0f, 256);
            bn_stats<<<512, 128, 0, stream>>>(xg, 256, STATS, N);
            bn_apply<<<(N * 128 + 255) / 256, 256, 0, stream>>>(xg, 256, STATS, cvp(ci, 4), cvp(ci, 5), N);
        }
        // joint (reused every stage): h = relu([x1,x2] @ jw1 + jb1) @ jw2 + jb2
        gemm(XCAT, 256, WB + P_J1, j_b1, TMP1, 128, N, 256, nullptr, 1);
        gemm(TMP1, 128, WB + P_J2, j_b2, FEATS + (st + 1) * 128, 640, N, 128, nullptr, 0);
    }

    // head: gather rows of [N x 640] feats, 3 relu GEMMs, then 128->2 + log_softmax
    gemm(FEATS, 640, WB + P_L1, l_b1, TMP1, 128, M, 640, idxs, 1);
    gemm(TMP1, 128, WB + P_L2, l_b2, AGG, 128, M, 128, nullptr, 1);
    gemm(AGG, 128, WB + P_L3, l_b3, TMP1, 128, M, 128, nullptr, 1);
    final_head<<<(M + 127) / 128, 128, 0, stream>>>(TMP1, l_w4, l_b4, (float*)d_out, M);
}
